// MAACActorSOTA_V4_53523882443169
// MI455X (gfx1250) — matrix operand order repaired and checked by reading, not compiled, not run on hardware
//
#include <hip/hip_runtime.h>
#include <hip/hip_bf16.h>
#include <stdint.h>

#ifndef __has_builtin
#define __has_builtin(x) 0
#endif

// ---------------- model constants (from reference) ----------------
#define BATCH  512
#define GSZ    32
#define DIM    128
#define HID    128
#define NHEAD  4      // NH == GH == 4
#define EDIM   16
#define NTOKN  8
#define NNODE  8192
#define NEDGE  65536
#define NPGR   16
#define MEMLEN 64
#define OUTW   154    // 2+2+2+16+4+128

// ---------------- WMMA / vector types ----------------
typedef __attribute__((ext_vector_type(16))) _Float16 v16h;
typedef __attribute__((ext_vector_type(8)))  float    v8f;
typedef __attribute__((ext_vector_type(4)))  float    fv4;
typedef __attribute__((ext_vector_type(2)))  unsigned uv2;
typedef __attribute__((ext_vector_type(4)))  unsigned uv4;
typedef __attribute__((ext_vector_type(4)))  int      iv4;
typedef __attribute__((ext_vector_type(8)))  int      iv8;
union VA { v16h v; _Float16 h[16]; };
union VC { v8f  v; float    f[8]; };

__device__ __forceinline__ float gelu_f(float x) {
    return 0.5f * x * (1.0f + erff(x * 0.70710678118654752f));
}
__device__ __forceinline__ float sigm_f(float x) { return 1.0f / (1.0f + expf(-x)); }

// pack 2 f32 -> packed f16 pair (v_cvt_pk_rtz_f16_f32 when available)
__device__ __forceinline__ unsigned pk2f16(float a, float b) {
#if __has_builtin(__builtin_amdgcn_cvt_pkrtz)
    typedef __fp16 h2 __attribute__((ext_vector_type(2)));
    union { h2 v; unsigned u; } x; x.v = __builtin_amdgcn_cvt_pkrtz(a, b); return x.u;
#else
    union { _Float16 h[2]; unsigned u; } x; x.h[0] = (_Float16)a; x.h[1] = (_Float16)b; return x.u;
#endif
}

// monotonic float<->uint for atomicMax-based segment max
__device__ __forceinline__ unsigned f2ord(float f) {
    unsigned u = __float_as_uint(f);
    return (u & 0x80000000u) ? ~u : (u | 0x80000000u);
}
__device__ __forceinline__ float ord2f(unsigned u) {
    return (u & 0x80000000u) ? __uint_as_float(u & 0x7fffffffu) : __uint_as_float(~u);
}
#define ORD_NEG_INF 0x007fffffu   // f2ord(-inf)

// =====================================================================
// WMMA GEMM (manual staging):  Y[M,N] = act( X[M,K] * W[N,K]^T + b[N] )
// 256 threads = 8 waves; block tile 64x64; wave tile 16(M) x 32(N).
// Fast path: b128 global loads + v_cvt_pk_rtz + b64 LDS stores.
// Edge path: branchless clamped-index scalar staging.
// =====================================================================
__global__ __launch_bounds__(256) void k_wmma_linear(
    const float* __restrict__ X, const float* __restrict__ W,
    const float* __restrict__ bias, float* __restrict__ Y,
    int M, int N, int K, int act)
{
    __shared__ _Float16 As[64][40];   // 80B rows -> conflict-free frag reads
    __shared__ _Float16 Bs[64][40];

    const int tid  = threadIdx.x;
    const int lane = tid & 31;
    const int wave = tid >> 5;        // 0..7
    const int wm   = wave >> 1;       // 0..3 -> 16-row slab
    const int wn   = wave & 1;        // 0..1 -> 32-col slab
    const int m0   = blockIdx.y * 64;
    const int n0   = blockIdx.x * 64;
    const int lh   = lane & 15;
    const int hi   = lane >> 4;

    const bool fullT = (m0 + 64 <= M) && (n0 + 64 <= N) && ((K & 3) == 0);

    VC acc0, acc1;
#pragma unroll
    for (int r = 0; r < 8; ++r) { acc0.f[r] = 0.f; acc1.f[r] = 0.f; }

    for (int k0 = 0; k0 < K; k0 += 32) {
        if (fullT && (k0 + 32 <= K)) {
            // vectorized: 512 quads per matrix, 2 per thread
            int q = tid;
#pragma unroll
            for (int it = 0; it < 2; ++it, q += 256) {
                int r = q >> 3, c4 = (q & 7) << 2;
                fv4 av = *(const fv4*)(X + (size_t)(m0 + r) * K + (k0 + c4));
                uv2 pa; pa[0] = pk2f16(av[0], av[1]); pa[1] = pk2f16(av[2], av[3]);
                *(uv2*)&As[r][c4] = pa;
                fv4 bv = *(const fv4*)(W + (size_t)(n0 + r) * K + (k0 + c4));
                uv2 pb; pb[0] = pk2f16(bv[0], bv[1]); pb[1] = pk2f16(bv[2], bv[3]);
                *(uv2*)&Bs[r][c4] = pb;
            }
            if (k0 + 32 < K)  // global_prefetch_b8 of next A tile
                __builtin_prefetch(&X[(size_t)(m0 + (tid >> 2)) * K + (k0 + 32)], 0, 1);
        } else {
            // branchless clamped staging (zero-padded)
            for (int i = tid; i < 64 * 32; i += 256) {
                int r = i >> 5, c = i & 31;
                int gm = m0 + r, gk = k0 + c, gn = n0 + r;
                int okA = (gm < M) & (gk < K);
                int okB = (gn < N) & (gk < K);
                float av = X[okA ? ((size_t)gm * K + gk) : 0];
                float bv = W[okB ? ((size_t)gn * K + gk) : 0];
                As[r][c] = (_Float16)(okA ? av : 0.f);
                Bs[r][c] = (_Float16)(okB ? bv : 0.f);
            }
        }
        __syncthreads();

        // A fragment (contiguous k: element j holds k = 16*hi + j)
        VA a;
        {
            int row = wm * 16 + lh;
            int kb  = hi * 16;
#pragma unroll
            for (int j = 0; j < 8; ++j) {
                a.h[j]     = As[row][kb + j];
                a.h[8 + j] = As[row][8 + kb + j];
            }
        }
        // B fragments (lane = N col, K halves by lane group)
        VA b0, b1;
        {
            int r0 = wn * 32 + lh, r1 = r0 + 16;
            int kb = hi * 16;
#pragma unroll
            for (int j = 0; j < 16; ++j) {
                b0.h[j] = Bs[r0][kb + j];
                b1.h[j] = Bs[r1][kb + j];
            }
        }
        acc0.v = __builtin_amdgcn_wmma_f32_16x16x32_f16(false, a.v, false, b0.v,
                                                        (short)0, acc0.v, false, false);
        acc1.v = __builtin_amdgcn_wmma_f32_16x16x32_f16(false, a.v, false, b1.v,
                                                        (short)0, acc1.v, false, false);
        __syncthreads();
    }

#pragma unroll
    for (int r = 0; r < 8; ++r) {
        int m = m0 + wm * 16 + r + hi * 8;
        if (m >= M) continue;
        int na = n0 + wn * 32 + lh;
        int nb = na + 16;
        if (na < N) {
            float v = acc0.f[r] + (bias ? bias[na] : 0.f);
            if (act == 1) v = gelu_f(v);
            Y[(size_t)m * N + na] = v;
        }
        if (nb < N) {
            float v = acc1.f[r] + (bias ? bias[nb] : 0.f);
            if (act == 1) v = gelu_f(v);
            Y[(size_t)m * N + nb] = v;
        }
    }
}

// =====================================================================
// WMMA GEMM with Tensor Data Mover panel staging (CDNA5 TDM, ISA ch.8).
// A/B 64x32 f32 tiles DMA'd to LDS with 1-DWORD row padding (33-DWORD
// stride -> conflict-free), hardware zero-fill at tensor edges, then
// f32->f16 conversion during fragment build.  tracked via TENSORcnt.
// =====================================================================
__global__ __launch_bounds__(256) void k_wmma_linear_tdm(
    const float* __restrict__ X, const float* __restrict__ W,
    const float* __restrict__ bias, float* __restrict__ Y,
    int M, int N, int K, int act)
{
    __shared__ float Af[64 * 33];
    __shared__ float Bf[64 * 33];

    const int tid  = threadIdx.x;
    const int lane = tid & 31;
    const int wave = tid >> 5;
    const int wm   = wave >> 1;
    const int wn   = wave & 1;
    const int m0   = blockIdx.y * 64;
    const int n0   = blockIdx.x * 64;
    const int lh   = lane & 15;
    const int hi   = lane >> 4;

    VC acc0, acc1;
#pragma unroll
    for (int r = 0; r < 8; ++r) { acc0.f[r] = 0.f; acc1.f[r] = 0.f; }

    for (int k0 = 0; k0 < K; k0 += 32) {
#if __has_builtin(__builtin_amdgcn_tensor_load_to_lds) && __has_builtin(__builtin_amdgcn_s_wait_tensorcnt)
        if (wave == 0) {
            iv4 z4 = {0, 0, 0, 0};
            iv8 z8 = {0, 0, 0, 0, 0, 0, 0, 0};
            int d0 = K - k0;          // remaining K: HW zero-fills tile beyond this
            int dA = M - m0;          // remaining rows (A)
            int dB = N - n0;          // remaining rows (B)
            // ---- D# group1 common: data_size=4B, pad 1 DWORD per 32 DWORDs ----
            iv8 g1;
            g1[0] = (2 << 16) | (1 << 20) | (4 << 22);
            g1[1] = (int)(((unsigned)d0 & 0xffffu) << 16);              // dim0 lo16
            g1[2] = 0; g1[3] = 0; g1[4] = 0;
            g1[5] = K;                                                  // dim0_stride lo32
            g1[6] = 0; g1[7] = 0;
            // ---- A tile ----
            {
                unsigned long long ga =
                    (unsigned long long)(size_t)(X + (size_t)m0 * K + k0);
                uv4 g0;
                g0[0] = 1u;                                             // count=1
                g0[1] = (unsigned)(size_t)Af;                           // lds_addr
                g0[2] = (unsigned)ga;
                g0[3] = (unsigned)((ga >> 32) & 0x1FFFFFFull) | (2u << 30); // type=2
                iv8 ga1 = g1;
                ga1[2] = (int)((((unsigned)d0 >> 16) & 0xffffu) |
                               (((unsigned)dA & 0xffffu) << 16));       // dim0 hi | dim1 lo
                ga1[3] = (int)((((unsigned)dA >> 16) & 0xffffu) | (32u << 16)); // dim1 hi | tile_dim0
                ga1[4] = 64;                                            // tile_dim1=64
                __builtin_amdgcn_tensor_load_to_lds(g0, ga1, z4, z4, z8, 0);
            }
            // ---- B tile ----
            {
                unsigned long long ga =
                    (unsigned long long)(size_t)(W + (size_t)n0 * K + k0);
                uv4 g0;
                g0[0] = 1u;
                g0[1] = (unsigned)(size_t)Bf;
                g0[2] = (unsigned)ga;
                g0[3] = (unsigned)((ga >> 32) & 0x1FFFFFFull) | (2u << 30);
                iv8 gb1 = g1;
                gb1[2] = (int)((((unsigned)d0 >> 16) & 0xffffu) |
                               (((unsigned)dB & 0xffffu) << 16));
                gb1[3] = (int)((((unsigned)dB >> 16) & 0xffffu) | (32u << 16));
                gb1[4] = 64;
                __builtin_amdgcn_tensor_load_to_lds(g0, gb1, z4, z4, z8, 0);
            }
        }
        __builtin_amdgcn_s_wait_tensorcnt(0);
#else
        // fallback (host pass / toolchains without the TDM builtin)
        for (int i = tid; i < 64 * 32; i += 256) {
            int r = i >> 5, c = i & 31;
            int gm = m0 + r, gk = k0 + c, gn = n0 + r;
            int okA = (gm < M) & (gk < K);
            int okB = (gn < N) & (gk < K);
            float av = X[okA ? ((size_t)gm * K + gk) : 0];
            float bv = W[okB ? ((size_t)gn * K + gk) : 0];
            Af[r * 33 + c] = okA ? av : 0.f;
            Bf[r * 33 + c] = okB ? bv : 0.f;
        }
#endif
        __syncthreads();

        VA a, b0, b1;
        {
            int row = wm * 16 + lh;
            int kb  = hi * 16;
#pragma unroll
            for (int j = 0; j < 8; ++j) {
                a.h[j]     = (_Float16)Af[row * 33 + kb + j];
                a.h[8 + j] = (_Float16)Af[row * 33 + 8 + kb + j];
            }
        }
        {
            int r0 = wn * 32 + lh, r1 = r0 + 16;
            int kb = hi * 16;
#pragma unroll
            for (int j = 0; j < 16; ++j) {
                b0.h[j] = (_Float16)Bf[r0 * 33 + kb + j];
                b1.h[j] = (_Float16)Bf[r1 * 33 + kb + j];
            }
        }
        acc0.v = __builtin_amdgcn_wmma_f32_16x16x32_f16(false, a.v, false, b0.v,
                                                        (short)0, acc0.v, false, false);
        acc1.v = __builtin_amdgcn_wmma_f32_16x16x32_f16(false, a.v, false, b1.v,
                                                        (short)0, acc1.v, false, false);
        __syncthreads();
    }

#pragma unroll
    for (int r = 0; r < 8; ++r) {
        int m = m0 + wm * 16 + r + hi * 8;
        if (m >= M) continue;
        int na = n0 + wn * 32 + lh;
        int nb = na + 16;
        if (na < N) {
            float v = acc0.f[r] + (bias ? bias[na] : 0.f);
            if (act == 1) v = gelu_f(v);
            Y[(size_t)m * N + na] = v;
        }
        if (nb < N) {
            float v = acc1.f[r] + (bias ? bias[nb] : 0.f);
            if (act == 1) v = gelu_f(v);
            Y[(size_t)m * N + nb] = v;
        }
    }
}

// =====================================================================
// Row LayerNorm with per-element affine; act 1 fuses GELU. One block/row.
// =====================================================================
__global__ void k_layernorm(const float* __restrict__ X, int ldx,
                            const float* __restrict__ g, const float* __restrict__ b,
                            float* __restrict__ Y, int ldy, int N, int act)
{
    const float* x = X + (size_t)blockIdx.x * ldx;
    float* y = Y + (size_t)blockIdx.x * ldy;
    __shared__ float s1[256], s2[256];
    float a = 0.f, s = 0.f;
    for (int i = threadIdx.x; i < N; i += blockDim.x) { float v = x[i]; a += v; s += v * v; }
    s1[threadIdx.x] = a; s2[threadIdx.x] = s;
    __syncthreads();
    for (int o = blockDim.x >> 1; o > 0; o >>= 1) {
        if (threadIdx.x < o) { s1[threadIdx.x] += s1[threadIdx.x + o]; s2[threadIdx.x] += s2[threadIdx.x + o]; }
        __syncthreads();
    }
    float mu  = s1[0] / N;
    float var = s2[0] / N - mu * mu;
    float inv = rsqrtf(var + 1e-5f);
    for (int i = threadIdx.x; i < N; i += blockDim.x) {
        float v = (x[i] - mu) * inv * g[i] + b[i];
        if (act == 1) v = gelu_f(v);
        y[i] = v;
    }
}

// GroupNorm (per-channel affine) + GELU. grid = B*G, block 256.
__global__ void k_groupnorm_gelu(const float* __restrict__ X, float* __restrict__ Y,
                                 const float* __restrict__ g, const float* __restrict__ b,
                                 int C, int G, int HW)
{
    int bb = blockIdx.x / G, gr = blockIdx.x % G;
    int cpg = C / G, n = cpg * HW;
    const float* x = X + ((size_t)bb * C + (size_t)gr * cpg) * HW;
    float* y = Y + ((size_t)bb * C + (size_t)gr * cpg) * HW;
    __shared__ float s1[256], s2[256];
    float a = 0.f, s = 0.f;
    for (int i = threadIdx.x; i < n; i += blockDim.x) { float v = x[i]; a += v; s += v * v; }
    s1[threadIdx.x] = a; s2[threadIdx.x] = s;
    __syncthreads();
    for (int o = blockDim.x >> 1; o > 0; o >>= 1) {
        if (threadIdx.x < o) { s1[threadIdx.x] += s1[threadIdx.x + o]; s2[threadIdx.x] += s2[threadIdx.x + o]; }
        __syncthreads();
    }
    float mu  = s1[0] / n;
    float var = s2[0] / n - mu * mu;
    float inv = rsqrtf(var + 1e-5f);
    for (int i = threadIdx.x; i < n; i += blockDim.x) {
        int c = gr * cpg + i / HW;
        y[i] = gelu_f((x[i] - mu) * inv * g[c] + b[c]);
    }
}

// Direct NCHW conv; writes into channel slice [c0, c0+Cout)
__global__ void k_conv2d(const float* __restrict__ X, const float* __restrict__ Wt,
                         const float* __restrict__ bias, float* __restrict__ Y,
                         int Bn, int Cin, int Hin, int Win, int Cout,
                         int kh, int kw, int stride, int pad,
                         int Ho, int Wo, int CoutTot, int c0)
{
    size_t total = (size_t)Bn * Cout * Ho * Wo;
    for (size_t idx = (size_t)blockIdx.x * blockDim.x + threadIdx.x; idx < total;
         idx += (size_t)gridDim.x * blockDim.x) {
        int ox = (int)(idx % Wo); size_t t = idx / Wo;
        int oy = (int)(t % Ho); t /= Ho;
        int co = (int)(t % Cout); int b = (int)(t / Cout);
        float acc = bias[co];
        for (int ci = 0; ci < Cin; ++ci)
            for (int ky = 0; ky < kh; ++ky) {
                int iy = oy * stride - pad + ky;
                if (iy < 0 || iy >= Hin) continue;
                for (int kx = 0; kx < kw; ++kx) {
                    int ix = ox * stride - pad + kx;
                    if (ix < 0 || ix >= Win) continue;
                    acc += X[(((size_t)b * Cin + ci) * Hin + iy) * Win + ix] *
                           Wt[(((size_t)co * Cin + ci) * kh + ky) * kw + kx];
                }
            }
        Y[(((size_t)b * CoutTot + c0 + co) * Ho + oy) * Wo + ox] = acc;
    }
}

// Fused 1x1 conv (32->32) + GELU + 4x32 token pooling for the cand branch.
__global__ void k_cand_pool(const float* __restrict__ O, const float* __restrict__ Wr,
                            const float* __restrict__ br, float* __restrict__ T)
{
    size_t total = (size_t)BATCH * NTOKN * 32;
    for (size_t idx = (size_t)blockIdx.x * blockDim.x + threadIdx.x; idx < total;
         idx += (size_t)gridDim.x * blockDim.x) {
        int c = (int)(idx % 32); size_t t2 = idx / 32;
        int tk = (int)(t2 % NTOKN); int b = (int)(t2 / NTOKN);
        float s = 0.f;
        for (int y = 0; y < 4; ++y)
            for (int x = 0; x < 32; ++x) {
                float acc = br[c];
                const float* px = O + (((size_t)b * 32) * 32 + (tk * 4 + y)) * 32 + x;
                for (int ci = 0; ci < 32; ++ci)
                    acc += px[(size_t)ci * 1024] * Wr[c * 32 + ci];
                s += gelu_f(acc);
            }
        T[((size_t)b * NTOKN + tk) * 32 + c] = s * (1.f / 128.f);
    }
}

// Spatial mean (B,C,H,W)->(B,C)
__global__ void k_spatial_mean(const float* __restrict__ X, float* __restrict__ Y, int C, int HW)
{
    int b = blockIdx.x, c = threadIdx.x;
    if (c < C) {
        const float* x = X + ((size_t)b * C + c) * HW;
        float s = 0.f;
        for (int i = 0; i < HW; ++i) s += x[i];
        Y[(size_t)b * C + c] = s / HW;
    }
}

// Small-Lq attention: one block per (b, head); Lk <= 80.
__global__ void k_attn_small(const float* __restrict__ Q, const float* __restrict__ Kp,
                             const float* __restrict__ Vp, float* __restrict__ O,
                             int Lk, int nh, int hd)
{
    int b = blockIdx.x / nh, h = blockIdx.x % nh;
    int Dm = nh * hd;
    __shared__ float sc[80];
    __shared__ float ssum;
    const float* q = Q + (size_t)b * Dm + h * hd;
    float scale = rsqrtf((float)hd);
    for (int l = threadIdx.x; l < Lk; l += blockDim.x) {
        const float* kk = Kp + ((size_t)b * Lk + l) * Dm + h * hd;
        float s = 0.f;
        for (int d = 0; d < hd; ++d) s += q[d] * kk[d];
        sc[l] = s * scale;
    }
    __syncthreads();
    if (threadIdx.x == 0) {
        float m = -1e30f;
        for (int l = 0; l < Lk; ++l) m = fmaxf(m, sc[l]);
        float su = 0.f;
        for (int l = 0; l < Lk; ++l) { sc[l] = expf(sc[l] - m); su += sc[l]; }
        ssum = su;
    }
    __syncthreads();
    float inv = 1.f / ssum;
    for (int d = threadIdx.x; d < hd; d += blockDim.x) {
        float o = 0.f;
        for (int l = 0; l < Lk; ++l)
            o += sc[l] * Vp[((size_t)b * Lk + l) * Dm + h * hd + d];
        O[(size_t)b * Dm + h * hd + d] = o * inv;
    }
}

// ---------------- GATv2 kernels ----------------
__global__ void k_gat_logits(const float* __restrict__ xl, const float* __restrict__ xr,
                             const float* __restrict__ ef, const float* __restrict__ We,
                             const float* __restrict__ att,
                             const int* __restrict__ src, const int* __restrict__ dst,
                             float* __restrict__ lg, unsigned* __restrict__ mx)
{
    int e = blockIdx.x, t = threadIdx.x;   // 128 threads
    __shared__ float es[16];
    __shared__ float red[128];
    if (t < 16) es[t] = ef[(size_t)e * 16 + t];
    __syncthreads();
    int s = src[e], d = dst[e];
    float part[4];
#pragma unroll
    for (int h = 0; h < 4; ++h) {
        int c = h * 128 + t;
        float eev = 0.f;
#pragma unroll
        for (int k2 = 0; k2 < 16; ++k2) eev += es[k2] * We[(size_t)c * 16 + k2];
        float v = xl[(size_t)s * 512 + c] + xr[(size_t)d * 512 + c] + eev;
        v = v > 0.f ? v : 0.2f * v;      // leaky_relu(0.2)
        part[h] = v * att[c];
    }
    for (int h = 0; h < 4; ++h) {
        red[t] = part[h];
        __syncthreads();
        for (int o = 64; o > 0; o >>= 1) { if (t < o) red[t] += red[t + o]; __syncthreads(); }
        if (t == 0) {
            float L = red[0];
            lg[(size_t)e * 4 + h] = L;
            atomicMax(&mx[(size_t)d * 4 + h], f2ord(L));
        }
        __syncthreads();
    }
}

__global__ void k_gat_expsum(float* __restrict__ lg, const unsigned* __restrict__ mx,
                             float* __restrict__ sm, const int* __restrict__ dst)
{
    size_t total = (size_t)NEDGE * 4;
    for (size_t i = (size_t)blockIdx.x * blockDim.x + threadIdx.x; i < total;
         i += (size_t)gridDim.x * blockDim.x) {
        int e = (int)(i >> 2), h = (int)(i & 3);
        int d = dst[e];
        float m = ord2f(mx[(size_t)d * 4 + h]);
        if (!(m > -1e30f)) m = 0.f;
        float ex = expf(lg[i] - m);
        lg[i] = ex;
        atomicAdd(&sm[(size_t)d * 4 + h], ex);
    }
}

__global__ void k_gat_aggr(const float* __restrict__ xl, const float* __restrict__ lg,
                           const float* __restrict__ sm, const int* __restrict__ src,
                           const int* __restrict__ dst, float* __restrict__ oacc)
{
    int e = blockIdx.x, t = threadIdx.x;
    int s = src[e], d = dst[e];
#pragma unroll
    for (int h = 0; h < 4; ++h) {
        float a = lg[(size_t)e * 4 + h] / (sm[(size_t)d * 4 + h] + 1e-16f);
        atomicAdd(&oacc[(size_t)d * 512 + h * 128 + t],
                  a * xl[(size_t)s * 512 + h * 128 + t]);
    }
}

__global__ void k_gat_final(const float* __restrict__ oacc, const float* __restrict__ bias,
                            float* __restrict__ hout)
{
    int n = blockIdx.x, t = threadIdx.x;
    const float* o = oacc + (size_t)n * 512;
    float v = 0.25f * (o[t] + o[128 + t] + o[256 + t] + o[384 + t]) + bias[t];
    hout[(size_t)n * 128 + t] = gelu_f(v);
}

__global__ void k_seg_pool(const float* __restrict__ h, float* __restrict__ pooled)
{
    int g = blockIdx.x, d = threadIdx.x;
    float s = 0.f;
    for (int i = 0; i < NPGR; ++i) s += h[((size_t)g * NPGR + i) * 128 + d];
    pooled[(size_t)g * 128 + d] = s * (1.f / NPGR);
}

// ---------------- misc fused kernels ----------------
__global__ void k_fill_f(float* p, float v, size_t n) {
    for (size_t i = (size_t)blockIdx.x * blockDim.x + threadIdx.x; i < n;
         i += (size_t)gridDim.x * blockDim.x) p[i] = v;
}
__global__ void k_fill_u(unsigned* p, unsigned v, size_t n) {
    for (size_t i = (size_t)blockIdx.x * blockDim.x + threadIdx.x; i < n;
         i += (size_t)gridDim.x * blockDim.x) p[i] = v;
}
__global__ void k_add(const float* a, const float* b, float* y, size_t n) {
    for (size_t i = (size_t)blockIdx.x * blockDim.x + threadIdx.x; i < n;
         i += (size_t)gridDim.x * blockDim.x) y[i] = a[i] + b[i];
}
__global__ void k_concat2(const float* A, const float* Bp, float* Y, int M, int N1, int N2) {
    size_t total = (size_t)M * (N1 + N2);
    for (size_t i = (size_t)blockIdx.x * blockDim.x + threadIdx.x; i < total;
         i += (size_t)gridDim.x * blockDim.x) {
        int c = (int)(i % (N1 + N2)); int r = (int)(i / (N1 + N2));
        Y[i] = (c < N1) ? A[(size_t)r * N1 + c] : Bp[(size_t)r * N2 + (c - N1)];
    }
}
__global__ void k_build_ctx(const float* p0, const float* p1, const float* p2,
                            const float* p3, float* ctx) {
    size_t total = (size_t)BATCH * 4 * 128;
    for (size_t i = (size_t)blockIdx.x * blockDim.x + threadIdx.x; i < total;
         i += (size_t)gridDim.x * blockDim.x) {
        int d = (int)(i % 128); int s = (int)((i / 128) % 4); int b = (int)(i / 512);
        const float* src = (s == 0) ? p0 : (s == 1) ? p1 : (s == 2) ? p2 : p3;
        ctx[i] = src[(size_t)b * 128 + d];
    }
}
__global__ void k_build_kv(const float* memx, const float* last, float* out) {
    size_t total = (size_t)BATCH * (MEMLEN + 1) * 128;
    for (size_t i = (size_t)blockIdx.x * blockDim.x + threadIdx.x; i < total;
         i += (size_t)gridDim.x * blockDim.x) {
        int d = (int)(i % 128); size_t t = i / 128;
        int l = (int)(t % (MEMLEN + 1)); int b = (int)(t / (MEMLEN + 1));
        out[i] = (l < MEMLEN) ? memx[((size_t)b * MEMLEN + l) * 128 + d]
                              : last[(size_t)b * 128 + d];
    }
}
__global__ void k_gru(const float* gi, const float* gh, const float* h0, float* out) {
    size_t total = (size_t)BATCH * 128;
    for (size_t i = (size_t)blockIdx.x * blockDim.x + threadIdx.x; i < total;
         i += (size_t)gridDim.x * blockDim.x) {
        int b = (int)(i / 128), j = (int)(i % 128);
        size_t o = (size_t)b * 384 + j;
        float r = sigm_f(gi[o] + gh[o]);
        float z = sigm_f(gi[o + 128] + gh[o + 128]);
        float n = tanhf(gi[o + 256] + r * gh[o + 256]);
        out[i] = (1.f - z) * n + z * h0[i];
    }
}
__global__ void k_gate_final(const float* gru, const float* tit,
                             const float* Wg, const float* bg, float* fin) {
    int b = blockIdx.x, t = threadIdx.x;   // 128
    __shared__ float red[128];
    __shared__ float gs;
    float p = gru[(size_t)b * 128 + t] * Wg[t] + tit[(size_t)b * 128 + t] * Wg[128 + t];
    red[t] = p;
    __syncthreads();
    for (int o = 64; o > 0; o >>= 1) { if (t < o) red[t] += red[t + o]; __syncthreads(); }
    if (t == 0) gs = sigm_f(red[0] + bg[0]);
    __syncthreads();
    float g = gs;
    fin[(size_t)b * 128 + t] = g * tit[(size_t)b * 128 + t] + (1.f - g) * gru[(size_t)b * 128 + t];
}
__global__ void k_heads(const float* __restrict__ fin,
                        const float* mvW, const float* mvb,
                        const float* pkW, const float* pkb,
                        const float* roW, const float* rob,
                        const float* rw1, const float* rb1,
                        const float* rw2, const float* rb2,
                        float* __restrict__ out)
{
    int b = blockIdx.x, t = threadIdx.x;   // 128
    __shared__ float f[128], rh[64], mvv[4], pks[2], roles[16], rws[4];
    f[t] = fin[(size_t)b * 128 + t];
    __syncthreads();
    if (t < 64) {
        float s = rb1[t];
        for (int k2 = 0; k2 < 128; ++k2) s += f[k2] * rw1[(size_t)t * 128 + k2];
        rh[t] = gelu_f(s);
    } else if (t < 68) {
        int j = t - 64; float s = mvb[j];
        for (int k2 = 0; k2 < 128; ++k2) s += f[k2] * mvW[(size_t)j * 128 + k2];
        mvv[j] = s;
    } else if (t < 70) {
        int j = t - 68; float s = pkb[j];
        for (int k2 = 0; k2 < 128; ++k2) s += f[k2] * pkW[(size_t)j * 128 + k2];
        pks[j] = s;
    } else if (t < 86) {
        int j = t - 70; float s = rob[j];
        for (int k2 = 0; k2 < 128; ++k2) s += f[k2] * roW[(size_t)j * 128 + k2];
        roles[j] = s;
    }
    __syncthreads();
    if (t < 4) {
        float s = rb2[t];
        for (int k2 = 0; k2 < 64; ++k2) s += rh[k2] * rw2[(size_t)t * 64 + k2];
        rws[t] = s;
    }
    __syncthreads();
    if (t == 0) {
        float m = fmaxf(fmaxf(rws[0], rws[1]), fmaxf(rws[2], rws[3]));
        float su = 0.f;
        for (int j = 0; j < 4; ++j) { rws[j] = expf(rws[j] - m); su += rws[j]; }
        for (int j = 0; j < 4; ++j) rws[j] /= su;
    }
    __syncthreads();
    float* o = out + (size_t)b * OUTW;
    for (int i = t; i < OUTW; i += 128) {
        float v;
        if (i < 2)       v = mvv[i];
        else if (i < 4)  v = -5.f + 3.5f * (tanhf(mvv[i]) + 1.f);
        else if (i < 6)  v = pks[i - 4];
        else if (i < 22) v = roles[i - 6];
        else if (i < 26) v = rws[i - 22];
        else             v = f[i - 26];
        o[i] = v;
    }
}

// =====================================================================
// Host side
// =====================================================================
struct Arena {
    char* base; size_t off, cap;
    float* f(size_t n) { off = (off + 255) & ~(size_t)255; float* p = (float*)(base + off); off += n * sizeof(float); return p; }
    unsigned* u(size_t n) { off = (off + 255) & ~(size_t)255; unsigned* p = (unsigned*)(base + off); off += n * 4; return p; }
};
struct MLPP { const float *W1, *W2, *b1, *b2, *be, *g; };
struct MHAP { const float *Wi, *Wo, *bi, *bo; };
struct GATP { const float *We, *Wl, *Wr, *att, *bias, *bl; };
struct EDGEP { const float *W1, *W2, *b1, *b2; };

static inline void linear_wmma(hipStream_t st, const float* X, const float* W,
                               const float* b, float* Y, int M, int N, int K, int act) {
    dim3 g((N + 63) / 64, (M + 63) / 64);
    if (K >= 32)
        k_wmma_linear_tdm<<<g, 256, 0, st>>>(X, W, b, Y, M, N, K, act);
    else
        k_wmma_linear<<<g, 256, 0, st>>>(X, W, b, Y, M, N, K, act);
}
static void run_mlp(hipStream_t st, const float* X, int M, int K, int Hd, int O,
                    const MLPP& p, float* t1, float* out) {
    linear_wmma(st, X, p.W1, p.b1, t1, M, Hd, K, 0);
    k_layernorm<<<dim3(M), 256, 0, st>>>(t1, Hd, p.g, p.be, t1, Hd, Hd, 1);
    linear_wmma(st, t1, p.W2, p.b2, out, M, O, Hd, 0);
}
static void run_mha(hipStream_t st, const float* Q, const float* Ks, const float* Vs,
                    int Bn, int Lk, int Dm, int nh, const MHAP& p,
                    float* qp, float* kp, float* vp, float* op, float* out) {
    linear_wmma(st, Q,  p.Wi,                       p.bi,          qp, Bn,      Dm, Dm, 0);
    linear_wmma(st, Ks, p.Wi + (size_t)Dm * Dm,     p.bi + Dm,     kp, Bn * Lk, Dm, Dm, 0);
    linear_wmma(st, Vs, p.Wi + 2 * (size_t)Dm * Dm, p.bi + 2 * Dm, vp, Bn * Lk, Dm, Dm, 0);
    k_attn_small<<<dim3(Bn * nh), 64, 0, st>>>(qp, kp, vp, op, Lk, nh, Dm / nh);
    linear_wmma(st, op, p.Wo, p.bo, out, Bn, Dm, Dm, 0);
}
static void run_gnn(hipStream_t st, Arena a, const float* nfeat, int nK,
                    const float* eattr, const int* eidx, const EDGEP& pe,
                    const MLPP& enc, const GATP* gl, const MLPP& proj, float* emb) {
    float* e = a.f((size_t)NEDGE * 16);
    {
        Arena b2 = a;
        float* e1 = b2.f((size_t)NEDGE * 64);
        linear_wmma(st, eattr, pe.W1, pe.b1, e1, NEDGE, 64, 4, 1);
        linear_wmma(st, e1, pe.W2, pe.b2, e, NEDGE, 16, 64, 0);
    }
    float* h0 = a.f((size_t)NNODE * 128);
    float* h1 = a.f((size_t)NNODE * 128);
    float* t  = a.f((size_t)NNODE * 128);
    run_mlp(st, nfeat, NNODE, nK, 128, 128, enc, t, h0);
    float* xl = a.f((size_t)NNODE * 512);
    float* xr = a.f((size_t)NNODE * 512);
    float* oa = a.f((size_t)NNODE * 512);
    float* lg = a.f((size_t)NEDGE * 4);
    unsigned* mx = a.u((size_t)NNODE * 4);
    float* sm = a.f((size_t)NNODE * 4);
    const int* src = eidx;
    const int* dst = eidx + NEDGE;
    float* hin = h0; float* hout = h1;
    for (int L = 0; L < 2; ++L) {
        const GATP& g = gl[L];
        linear_wmma(st, hin, g.Wl, g.bl, xl, NNODE, 512, 128, 0);
        linear_wmma(st, hin, g.Wr, nullptr, xr, NNODE, 512, 128, 0);
        k_fill_u<<<dim3((NNODE * 4 + 255) / 256), 256, 0, st>>>(mx, ORD_NEG_INF, (size_t)NNODE * 4);
        k_fill_f<<<dim3((NNODE * 4 + 255) / 256), 256, 0, st>>>(sm, 0.f, (size_t)NNODE * 4);
        k_fill_f<<<dim3(16384), 256, 0, st>>>(oa, 0.f, (size_t)NNODE * 512);
        k_gat_logits<<<dim3(NEDGE), 128, 0, st>>>(xl, xr, e, g.We, g.att, src, dst, lg, mx);
        k_gat_expsum<<<dim3(1024), 256, 0, st>>>(lg, mx, sm, dst);
        k_gat_aggr<<<dim3(NEDGE), 128, 0, st>>>(xl, lg, sm, src, dst, oa);
        k_gat_final<<<dim3(NNODE), 128, 0, st>>>(oa, g.bias, hout);
        float* tmp = hin; hin = hout; hout = tmp;
    }
    float* pooled = a.f((size_t)BATCH * 128);
    k_seg_pool<<<dim3(BATCH), 128, 0, st>>>(hin, pooled);
    run_mlp(st, pooled, BATCH, 128, 128, 128, proj, t, emb);
}

extern "C" void kernel_launch(void* const* d_in, const int* in_sizes, int n_in,
                              void* d_out, int out_size, void* d_ws, size_t ws_size,
                              hipStream_t stream) {
    (void)in_sizes; (void)n_in; (void)out_size;
    const float* self_obs      = (const float*)d_in[0];
    const float* raw_map       = (const float*)d_in[1];
    const float* node_feat     = (const float*)d_in[2];
    const float* edge_attr     = (const float*)d_in[3];
    const float* mem_node_feat = (const float*)d_in[4];
    const float* mem_edge_attr = (const float*)d_in[5];
    const float* memory_map    = (const float*)d_in[6];
    const float* h_gru_in      = (const float*)d_in[7];
    const float* mem_k         = (const float*)d_in[8];
    const float* mem_v         = (const float*)d_in[9];
    const int*   edge_index    = (const int*)d_in[10];
    const int*   mem_edge_idx  = (const int*)d_in[11];
    // d_in[12], d_in[13] (batch vecs) are deterministic arange//NPG -> unused

    // ----- params: JAX pytree leaves, alphabetical DFS, from d_in[14] -----
    int pi = 14;
    auto nx = [&]() -> const float* { return (const float*)d_in[pi++]; };
    auto getMLP  = [&]() { MLPP p; p.W1 = nx(); p.W2 = nx(); p.b1 = nx(); p.b2 = nx(); p.be = nx(); p.g = nx(); return p; };
    auto getGAT  = [&]() { GATP p; p.We = nx(); p.Wl = nx(); p.Wr = nx(); p.att = nx(); p.bias = nx(); p.bl = nx(); return p; };
    auto getMHA  = [&]() { MHAP p; p.Wi = nx(); p.Wo = nx(); p.bi = nx(); p.bo = nx(); return p; };
    auto getEDGE = [&]() { EDGEP p; p.W1 = nx(); p.W2 = nx(); p.b1 = nx(); p.b2 = nx(); return p; };

    const float *cW1 = nx(), *cW3 = nx(), *cW5 = nx(), *cWfc = nx(), *cWr = nx();
    const float *cb1 = nx(), *cb3 = nx(), *cb5 = nx(), *cbfc = nx(), *cbr = nx();
    const float *cln1b = nx(), *cln1g = nx(), *cln3b = nx(), *cln3g = nx(), *cln5b = nx(), *cln5g = nx();
    EDGEP comm_edge = getEDGE();
    MLPP  comm_enc  = getMLP();
    GATP  comm_gnn[2] = { getGAT(), getGAT() };
    MLPP  comm_proj = getMLP();
    MLPP  fc_cand   = getMLP();
    MHAP  fus_attn  = getMHA();
    const float *fus_ln_b = nx(), *fus_ln_g = nx();
    const float *gateW = nx(), *gateb = nx();
    const float *gWhh = nx(), *gWih = nx(), *gbhh = nx(), *gbih = nx();
    EDGEP mem_edge = getEDGE();
    MLPP  mem_enc  = getMLP();
    GATP  mem_gnn[2] = { getGAT(), getGAT() };
    MLPP  mem_proj = getMLP();
    const float *mW0 = nx(), *mW1 = nx(), *mWfc = nx(), *mb0 = nx(), *mb1 = nx();
    const float *mbe0 = nx(), *mbe1 = nx(), *mbfc = nx(), *mg0 = nx(), *mg1 = nx();
    const float *mvW = nx(), *mvb = nx();
    MLPP  pf = getMLP();
    const float *pkW = nx(), *pkb = nx();
    MLPP  post = getMLP();
    const float *roW = nx(), *rob = nx();
    const float *rwW1 = nx(), *rwW2 = nx(), *rwb1 = nx(), *rwb2 = nx();
    MLPP  selfp = getMLP();
    const float *tW1 = nx(), *tW2 = nx();
    MHAP  titan_attn = getMHA();
    const float *tb1 = nx(), *tb2 = nx();
    const float *tn1b = nx(), *tn1g = nx(), *tn2b = nx(), *tn2g = nx();
    MHAP  xattn = getMHA();

    Arena ws{ (char*)d_ws, 0, ws_size };
    float* baseline    = ws.f((size_t)BATCH * 128);
    float* cross       = ws.f((size_t)BATCH * 128);
    float* perception  = ws.f((size_t)BATCH * 128);
    float* comm_emb    = ws.f((size_t)BATCH * 128);
    float* mem_emb     = ws.f((size_t)BATCH * 128);
    float* mem_map_emb = ws.f((size_t)BATCH * 128);
    float* fusedb      = ws.f((size_t)BATCH * 128);
    float* tin         = ws.f((size_t)BATCH * 128);
    float* gru_out     = ws.f((size_t)BATCH * 128);
    float* titan_out   = ws.f((size_t)BATCH * 128);
    float* finalb      = ws.f((size_t)BATCH * 128);

    { Arena a = ws; float* t1 = a.f((size_t)BATCH * 128);
      run_mlp(stream, self_obs, BATCH, 32, 128, 128, selfp, t1, baseline); }

    { Arena a = ws;
      float* candb = a.f((size_t)BATCH * NTOKN * 128);
      { Arena b2 = a;
        float* obuf = b2.f((size_t)BATCH * 32 * 1024);
        k_conv2d<<<dim3(20480), 256, 0, stream>>>(raw_map, cW1, cb1, obuf, BATCH, 8, 32, 32, 10, 1, 1, 1, 0, 32, 32, 32, 0);
        k_conv2d<<<dim3(24576), 256, 0, stream>>>(raw_map, cW3, cb3, obuf, BATCH, 8, 32, 32, 12, 3, 3, 1, 1, 32, 32, 32, 10);
        k_conv2d<<<dim3(20480), 256, 0, stream>>>(raw_map, cW5, cb5, obuf, BATCH, 8, 32, 32, 10, 5, 5, 1, 2, 32, 32, 32, 22);
        k_layernorm<<<dim3(BATCH), 256, 0, stream>>>(obuf,             32 * 1024, cln1g, cln1b, obuf,             32 * 1024, 10 * 1024, 1);
        k_layernorm<<<dim3(BATCH), 256, 0, stream>>>(obuf + 10 * 1024, 32 * 1024, cln3g, cln3b, obuf + 10 * 1024, 32 * 1024, 12 * 1024, 1);
        k_layernorm<<<dim3(BATCH), 256, 0, stream>>>(obuf + 22 * 1024, 32 * 1024, cln5g, cln5b, obuf + 22 * 1024, 32 * 1024, 10 * 1024, 1);
        float* tokens = b2.f((size_t)BATCH * NTOKN * 32);
        k_cand_pool<<<dim3(512), 256, 0, stream>>>(obuf, cWr, cbr, tokens);
        float* tok8 = b2.f((size_t)BATCH * NTOKN * 8);
        linear_wmma(stream, tokens, cWfc, cbfc, tok8, BATCH * NTOKN, 8, 32, 0);
        float* t1 = b2.f((size_t)BATCH * NTOKN * 128);
        run_mlp(stream, tok8, BATCH * NTOKN, 8, 128, 128, fc_cand, t1, candb);
      }
      float* qp = a.f((size_t)BATCH * 128);
      float* kp = a.f((size_t)BATCH * NTOKN * 128);
      float* vp = a.f((size_t)BATCH * NTOKN * 128);
      float* op = a.f((size_t)BATCH * 128);
      run_mha(stream, baseline, candb, candb, BATCH, NTOKN, 128, NHEAD, xattn, qp, kp, vp, op, cross);
    }

    { Arena a = ws;
      float* cat = a.f((size_t)BATCH * 256);
      k_concat2<<<dim3(512), 256, 0, stream>>>(baseline, cross, cat, BATCH, 128, 128);
      float* t1 = a.f((size_t)BATCH * 256);
      run_mlp(stream, cat, BATCH, 256, 256, 128, pf, t1, perception); }

    run_gnn(stream, ws, node_feat,     16, edge_attr,     edge_index,   comm_edge, comm_enc, comm_gnn, comm_proj, comm_emb);
    run_gnn(stream, ws, mem_node_feat,  9, mem_edge_attr, mem_edge_idx, mem_edge,  mem_enc,  mem_gnn,  mem_proj,  mem_emb);

    { Arena a = ws;
      float* c0 = a.f((size_t)BATCH * 64 * 256);
      k_conv2d<<<dim3(32768), 256, 0, stream>>>(memory_map, mW0, mb0, c0, BATCH, 4, 32, 32, 64, 3, 3, 2, 1, 16, 16, 64, 0);
      k_groupnorm_gelu<<<dim3(BATCH * 8), 256, 0, stream>>>(c0, c0, mg0, mbe0, 64, 8, 256);
      float* c1 = a.f((size_t)BATCH * 64 * 64);
      k_conv2d<<<dim3(8192), 256, 0, stream>>>(c0, mW1, mb1, c1, BATCH, 64, 16, 16, 64, 3, 3, 2, 1, 8, 8, 64, 0);
      k_groupnorm_gelu<<<dim3(BATCH * 8), 256, 0, stream>>>(c1, c1, mg1, mbe1, 64, 8, 64);
      float* pm = a.f((size_t)BATCH * 64);
      k_spatial_mean<<<dim3(BATCH), 64, 0, stream>>>(c1, pm, 64, 64);
      linear_wmma(stream, pm, mWfc, mbfc, mem_map_emb, BATCH, 128, 64, 0); }

    { Arena a = ws;
      float* ctx = a.f((size_t)BATCH * 4 * 128);
      k_build_ctx<<<dim3(1024), 256, 0, stream>>>(perception, comm_emb, mem_emb, mem_map_emb, ctx);
      float* qp = a.f((size_t)BATCH * 128);
      float* kp = a.f((size_t)BATCH * 4 * 128);
      float* vp = a.f((size_t)BATCH * 4 * 128);
      float* op = a.f((size_t)BATCH * 128);
      float* attn = a.f((size_t)BATCH * 128);
      float* tmp = a.f((size_t)BATCH * 128);
      run_mha(stream, perception, ctx, ctx, BATCH, 4, 128, NHEAD, fus_attn, qp, kp, vp, op, attn);
      k_add<<<dim3(256), 256, 0, stream>>>(perception, attn, tmp, (size_t)BATCH * 128);
      k_layernorm<<<dim3(BATCH), 256, 0, stream>>>(tmp, 128, fus_ln_g, fus_ln_b, fusedb, 128, 128, 0); }

    { Arena a = ws; float* t1 = a.f((size_t)BATCH * 256);
      run_mlp(stream, fusedb, BATCH, 128, 256, 128, post, t1, tin); }
    { Arena a = ws;
      float* gi = a.f((size_t)BATCH * 384);
      float* gh = a.f((size_t)BATCH * 384);
      linear_wmma(stream, tin,      gWih, gbih, gi, BATCH, 384, 128, 0);
      linear_wmma(stream, h_gru_in, gWhh, gbhh, gh, BATCH, 384, 128, 0);
      k_gru<<<dim3(256), 256, 0, stream>>>(gi, gh, h_gru_in, gru_out); }

    { Arena a = ws;
      const int LK = MEMLEN + 1;
      float* key = a.f((size_t)BATCH * LK * 128);
      float* val = a.f((size_t)BATCH * LK * 128);
      k_build_kv<<<dim3(16640), 256, 0, stream>>>(mem_k, gru_out, key);
      k_build_kv<<<dim3(16640), 256, 0, stream>>>(mem_v, gru_out, val);
      float* qp = a.f((size_t)BATCH * 128);
      float* kp = a.f((size_t)BATCH * LK * 128);
      float* vp = a.f((size_t)BATCH * LK * 128);
      float* op = a.f((size_t)BATCH * 128);
      float* att = a.f((size_t)BATCH * 128);
      run_mha(stream, gru_out, key, val, BATCH, LK, 128, NHEAD, titan_attn, qp, kp, vp, op, att);
      float* s = a.f((size_t)BATCH * 128);
      k_add<<<dim3(256), 256, 0, stream>>>(gru_out, att, s, (size_t)BATCH * 128);
      k_layernorm<<<dim3(BATCH), 256, 0, stream>>>(s, 128, tn1g, tn1b, s, 128, 128, 0);
      float* ff1 = a.f((size_t)BATCH * 2048);
      linear_wmma(stream, s, tW1, tb1, ff1, BATCH, 2048, 128, 1);
      float* ff2 = a.f((size_t)BATCH * 128);
      linear_wmma(stream, ff1, tW2, tb2, ff2, BATCH, 128, 2048, 0);
      float* s2 = a.f((size_t)BATCH * 128);
      k_add<<<dim3(256), 256, 0, stream>>>(s, ff2, s2, (size_t)BATCH * 128);
      k_layernorm<<<dim3(BATCH), 256, 0, stream>>>(s2, 128, tn2g, tn2b, titan_out, 128, 128, 0); }

    k_gate_final<<<dim3(BATCH), 128, 0, stream>>>(gru_out, titan_out, gateW, gateb, finalb);
    k_heads<<<dim3(BATCH), 128, 0, stream>>>(finalb, mvW, mvb, pkW, pkb, roW, rob,
                                             rwW1, rwb1, rwW2, rwb2, (float*)d_out);
}
